// MultiHeadAttention_72327249265063
// MI455X (gfx1250) — matrix operand order repaired and checked by reading, not compiled, not run on hardware
//
#include <hip/hip_runtime.h>

// MI455X (gfx1250) multi-head attention forward, v3.
//  - One-time fp32 -> f16 pack pass; all GEMM/attention inner loops load f16
//    directly (fragments = global_load_b128, no cvt chains on the WMMA path).
//  - All matmuls on v_wmma_f32_16x16x32_f16 (fp32 accumulate).
//  - V projected directly into transposed [B,H,DK,S] layout so the P@V
//    B-fragment is two b128 loads from one contiguous Vt row.
//  - Flash attention (online softmax): S x S scores never materialized.
//  - All problem dims are compile-time powers of two -> epilogue indexing is
//    shifts/masks (no integer-division expansions); GEMM mode is a template.
//  - Workspace (f16): q|k|v (8MB ea) + 4 weights (2MB ea) + Qh|Kh|Vt|attn
//    (8MB ea) = 64MB.

typedef __attribute__((ext_vector_type(16))) _Float16 v16h;
typedef __attribute__((ext_vector_type(8)))  _Float16 v8h;
typedef __attribute__((ext_vector_type(8)))  float    v8f;

constexpr int CB  = 2;      // batch
constexpr int CS  = 2048;   // sequence
constexpr int CD  = 1024;   // model dim
constexpr int CH  = 16;     // heads
constexpr int CDK = 64;     // head dim
constexpr int CM  = CB * CS;

#define WMMA_F16(a, b, c) \
  __builtin_amdgcn_wmma_f32_16x16x32_f16(false, (a), false, (b), (short)0, (c), false, false)

// reduce across a 16-lane half-wave (xor masks 1,2,4,8 stay inside the half)
__device__ __forceinline__ float half16_max(float v) {
#pragma unroll
  for (int m = 1; m < 16; m <<= 1) v = fmaxf(v, __shfl_xor(v, m, 32));
  return v;
}
__device__ __forceinline__ float half16_sum(float v) {
#pragma unroll
  for (int m = 1; m < 16; m <<= 1) v += __shfl_xor(v, m, 32);
  return v;
}

// ---------------------------------------------------------------------------
// fp32 -> f16 pack (n multiple of 8)
// ---------------------------------------------------------------------------
__global__ __launch_bounds__(256)
void pack_f16(const float* __restrict__ X, _Float16* __restrict__ Y, int n) {
  const int i = (blockIdx.x * 256 + threadIdx.x) * 8;
  if (i + 8 <= n) {
#pragma unroll
    for (int j = 0; j < 8; j++) Y[i + j] = (_Float16)X[i + j];
  }
}

// ---------------------------------------------------------------------------
// Y = X @ W^T + bias.  X:[M,K=CD] f16 row-major, W:[N=CD,K=CD] f16 row-major.
// One wave computes a 32x64 tile: 2 A frags x 4 B frags = 8 WMMA per k-step.
// MODE 0: f32 out [M,CD]; MODE 1: f16 out [B,H,S,DK]; MODE 2: f16 [B,H,DK,S].
// ---------------------------------------------------------------------------
template <int MODE>
__global__ __launch_bounds__(32)
void gemm_f16_wmma(const _Float16* __restrict__ X, const _Float16* __restrict__ W,
                   const float* __restrict__ bias, void* __restrict__ Yv) {
  constexpr int K = CD;
  const int lane  = threadIdx.x & 31;
  const int col   = lane & 15;
  const int hi    = lane >> 4;            // 0 | 1 (half-wave)
  const int nbase = blockIdx.x * 64;
  const int mbase = blockIdx.y * 32;

  v8f acc[2][4] = {};
  for (int k = 0; k < K; k += 32) {
    v16h a[2];
#pragma unroll
    for (int r = 0; r < 2; r++) {
      // A layout: lane<16 holds K {0..7,16..23}, lane>=16 holds {8..15,24..31}
      const _Float16* xp = X + (size_t)(mbase + r * 16 + col) * K + k + hi * 8;
      const v8h lo = *(const v8h*)xp;
      const v8h hh = *(const v8h*)(xp + 16);
#pragma unroll
      for (int j = 0; j < 8; j++) { a[r][j] = lo[j]; a[r][8 + j] = hh[j]; }
    }
#pragma unroll
    for (int nt = 0; nt < 4; nt++) {
      // B column = weight row: K {0..7,16..23} / {8..15,24..31} (two b128 loads)
      const _Float16* wp = W + (size_t)(nbase + nt * 16 + col) * K + k + hi * 8;
      const v8h blo = *(const v8h*)wp;
      const v8h bhh = *(const v8h*)(wp + 16);
      v16h b;
#pragma unroll
      for (int j = 0; j < 8; j++) { b[j] = blo[j]; b[8 + j] = bhh[j]; }
      acc[0][nt] = WMMA_F16(a[0], b, acc[0][nt]);
      acc[1][nt] = WMMA_F16(a[1], b, acc[1][nt]);
    }
  }

#pragma unroll
  for (int r = 0; r < 2; r++) {
#pragma unroll
    for (int nt = 0; nt < 4; nt++) {
      const int   n  = nbase + nt * 16 + col;
      const float bv = bias[n];
#pragma unroll
      for (int i = 0; i < 8; i++) {            // C: vgpr i <-> row i + 8*hi
        const int   m   = mbase + r * 16 + i + hi * 8;
        const float val = acc[r][nt][i] + bv;
        if (MODE == 0) {
          ((float*)Yv)[(size_t)m * CD + n] = val;
        } else {
          const int h  = n >> 6,  dk = n & (CDK - 1);   // CDK = 64
          const int bi = m >> 11, s  = m & (CS - 1);    // CS  = 2048
          _Float16* Yh = (_Float16*)Yv;
          if (MODE == 1) Yh[(((size_t)bi * CH + h) * CS + s) * CDK + dk] = (_Float16)val;
          else           Yh[(((size_t)bi * CH + h) * CDK + dk) * CS + s] = (_Float16)val;
        }
      }
    }
  }
}

// ---------------------------------------------------------------------------
// Causal flash attention, DK = 64. One wave per (b*H+h, 16-query tile).
// Qh,Kh: f16 [B,H,S,64]; Vt: f16 [B,H,64,S]; O: f16 [B,S,H*64].
// With S % 32 == 0, streamed key indices never exceed S-1 (mask handles the
// causal boundary), so there are no clamps / scalar paths.
// ---------------------------------------------------------------------------
__global__ __launch_bounds__(32)
void flash_attn_wmma(const _Float16* __restrict__ Qh, const _Float16* __restrict__ Kh,
                     const _Float16* __restrict__ Vt, _Float16* __restrict__ O) {
  const int bh   = blockIdx.x;            // b*H + h
  const int q0   = blockIdx.y * 16;
  const int lane = threadIdx.x & 31;
  const int col  = lane & 15;
  const int hi   = lane >> 4;

  const _Float16* Qp = Qh + (size_t)bh * CS * CDK;
  const _Float16* Kp = Kh + (size_t)bh * CS * CDK;
  const _Float16* Vp = Vt + (size_t)bh * CDK * CS;

  __shared__ _Float16 P[16][40];          // 16x32 probs tile (stride 40 = 80B)

  // Q fragments for the two K-chunks; fold 1/sqrt(64)=0.125 (exact in f16).
  const _Float16 qscale = (_Float16)0.125f;
  v16h qf[2];
  {
    const int row = q0 + col;
#pragma unroll
    for (int c = 0; c < 2; c++) {
      const _Float16* qp = Qp + (size_t)row * CDK + c * 32 + hi * 8;
      const v8h lo = *(const v8h*)qp;
      const v8h hh = *(const v8h*)(qp + 16);
#pragma unroll
      for (int j = 0; j < 8; j++) {
        qf[c][j]     = lo[j] * qscale;
        qf[c][8 + j] = hh[j] * qscale;
      }
    }
  }

  v8f   o[4] = {};
  float mrow[8], lrow[8];
#pragma unroll
  for (int i = 0; i < 8; i++) { mrow[i] = -1e30f; lrow[i] = 0.0f; }

  const int kend = q0 + 16;               // causal: keys < q0+16
  for (int t = 0; t < kend; t += 32) {
    if (t + 32 < kend) {                  // pull next tiles toward the caches
      __builtin_prefetch(Kp + (size_t)(t + 32 + col) * CDK, 0, 1);
      __builtin_prefetch(Vp + (size_t)(col * 4) * CS + t + 32, 0, 1);
    }

    // ---- scores: two 16-key subtiles, each accumulating 2 K-chunks ----
    v8f c[2] = {};
#pragma unroll
    for (int sub = 0; sub < 2; sub++) {
      const int key = t + sub * 16 + col;
#pragma unroll
      for (int kc = 0; kc < 2; kc++) {
        const _Float16* kp = Kp + (size_t)key * CDK + kc * 32 + hi * 8;
        const v8h klo = *(const v8h*)kp;
        const v8h khh = *(const v8h*)(kp + 16);
        v16h bf;
#pragma unroll
        for (int j = 0; j < 8; j++) { bf[j] = klo[j]; bf[8 + j] = khh[j]; }
        c[sub] = WMMA_F16(qf[kc], bf, c[sub]);
      }
#pragma unroll
      for (int i = 0; i < 8; i++) {       // causal mask: key > query -> -inf
        const int qi = q0 + i + hi * 8;
        if (key > qi) c[sub][i] = -1e30f;
      }
    }

    // ---- online softmax update (per output row) ----
    float alpha[8];
#pragma unroll
    for (int i = 0; i < 8; i++) {
      const float tmax = half16_max(fmaxf(c[0][i], c[1][i]));
      const float mnew = fmaxf(mrow[i], tmax);
      alpha[i] = __expf(mrow[i] - mnew);
      mrow[i]  = mnew;
      const float p0 = __expf(c[0][i] - mnew);
      const float p1 = __expf(c[1][i] - mnew);
      c[0][i] = p0; c[1][i] = p1;
      lrow[i] = lrow[i] * alpha[i] + half16_sum(p0 + p1);
    }

    // rescale the running output accumulator
#pragma unroll
    for (int nt = 0; nt < 4; nt++)
#pragma unroll
      for (int i = 0; i < 8; i++) o[nt][i] *= alpha[i];

    // ---- stage probs to LDS (C-layout -> row-major) ----
#pragma unroll
    for (int i = 0; i < 8; i++) {
      const int r = i + hi * 8;
      P[r][col]      = (_Float16)c[0][i];
      P[r][16 + col] = (_Float16)c[1][i];
    }
    // single wave: DS ops are in-order; compiler inserts s_wait_dscnt

    // ---- reload probs as a 16x32 A fragment (two ds b128 loads) ----
    v16h pf;
    {
      const _Float16* pp = &P[col][hi * 8];
      const v8h lo = *(const v8h*)pp;
      const v8h hh = *(const v8h*)(pp + 16);
#pragma unroll
      for (int j = 0; j < 8; j++) { pf[j] = lo[j]; pf[8 + j] = hh[j]; }
    }

    // ---- P @ V: 4 N-tiles over DK=64; Vt gives contiguous B columns ----
#pragma unroll
    for (int nt = 0; nt < 4; nt++) {
      const int n = nt * 16 + col;        // B column = dk index
      const _Float16* vp = Vp + (size_t)n * CS + t + hi * 8;
      const v8h vlo = *(const v8h*)vp;
      const v8h vhh = *(const v8h*)(vp + 16);
      v16h vf;
#pragma unroll
      for (int j = 0; j < 8; j++) { vf[j] = vlo[j]; vf[8 + j] = vhh[j]; }
      o[nt] = WMMA_F16(pf, vf, o[nt]);
    }
  }

  // ---- normalize and write out as f16 [B, S, H*DK] ----
  const int bi = bh >> 4, h = bh & (CH - 1);
#pragma unroll
  for (int i = 0; i < 8; i++) {
    const int   s   = q0 + i + hi * 8;
    const float inv = 1.0f / lrow[i];
#pragma unroll
    for (int nt = 0; nt < 4; nt++) {
      const int dk = nt * 16 + col;
      O[(((size_t)bi * CS + s) * CH + h) * CDK + dk] = (_Float16)(o[nt][i] * inv);
    }
  }
}

// ---------------------------------------------------------------------------
extern "C" void kernel_launch(void* const* d_in, const int* in_sizes, int n_in,
                              void* d_out, int out_size, void* d_ws, size_t ws_size,
                              hipStream_t stream) {
  const float* q    = (const float*)d_in[0];
  const float* k    = (const float*)d_in[1];
  const float* v    = (const float*)d_in[2];
  const float* wq_w = (const float*)d_in[3];
  const float* wq_b = (const float*)d_in[4];
  const float* wk_w = (const float*)d_in[5];
  const float* wk_b = (const float*)d_in[6];
  const float* wv_w = (const float*)d_in[7];
  const float* wv_b = (const float*)d_in[8];
  const float* wo_w = (const float*)d_in[9];
  const float* wo_b = (const float*)d_in[10];
  // d_in[11] = causal mask (tril) — causality applied analytically.

  const size_t A  = (size_t)CM * CD;          // 4,194,304 halfs per activation
  const size_t DD = (size_t)CD * CD;          // 1,048,576 halfs per weight

  _Float16* ws   = (_Float16*)d_ws;
  _Float16* qf   = ws;                        // packed activations
  _Float16* kf   = ws + A;
  _Float16* vf   = ws + 2 * A;
  _Float16* wq16 = ws + 3 * A;                // packed weights
  _Float16* wk16 = wq16 + DD;
  _Float16* wv16 = wq16 + 2 * DD;
  _Float16* wo16 = wq16 + 3 * DD;
  _Float16* Qh   = wq16 + 4 * DD;             // [B,H,S,DK]
  _Float16* Kh   = Qh + A;                    // [B,H,S,DK]
  _Float16* Vt   = Qh + 2 * A;                // [B,H,DK,S]
  _Float16* attn = Qh + 3 * A;                // [B,S,H*DK]
  // total: 7*A + 4*DD halfs = 64MB

  dim3 b256(256), b32(32);

  pack_f16<<<dim3(A / 2048),  b256, 0, stream>>>(q,    qf,   (int)A);
  pack_f16<<<dim3(A / 2048),  b256, 0, stream>>>(k,    kf,   (int)A);
  pack_f16<<<dim3(A / 2048),  b256, 0, stream>>>(v,    vf,   (int)A);
  pack_f16<<<dim3(DD / 2048), b256, 0, stream>>>(wq_w, wq16, (int)DD);
  pack_f16<<<dim3(DD / 2048), b256, 0, stream>>>(wk_w, wk16, (int)DD);
  pack_f16<<<dim3(DD / 2048), b256, 0, stream>>>(wv_w, wv16, (int)DD);
  pack_f16<<<dim3(DD / 2048), b256, 0, stream>>>(wo_w, wo16, (int)DD);

  dim3 gproj(CD / 64, CM / 32);               // 16 x 128 waves
  gemm_f16_wmma<1><<<gproj, b32, 0, stream>>>(qf, wq16, wq_b, Qh);
  gemm_f16_wmma<1><<<gproj, b32, 0, stream>>>(kf, wk16, wk_b, Kh);
  gemm_f16_wmma<2><<<gproj, b32, 0, stream>>>(vf, wv16, wv_b, Vt);

  dim3 gattn(CB * CH, CS / 16);               // 32 x 128 waves
  flash_attn_wmma<<<gattn, b32, 0, stream>>>(Qh, Kh, Vt, attn);

  gemm_f16_wmma<0><<<gproj, b32, 0, stream>>>(attn, wo16, wo_b, (float*)d_out);
}
